// Seq2Seq_48180943127080
// MI455X (gfx1250) — compile-verified, matrix operand order checked by reading, not run on hardware
//
#include <hip/hip_runtime.h>
#include <hip/hip_bf16.h>

#define Bv 64
#define Tv 64
#define Hv 512
#define Vv 20001
#define G3 1536                 // 3*H
#define PADTOK 20000
#define MR (Tv * Bv)            // 4096 rows (t*B+b)

typedef __attribute__((ext_vector_type(16))) __bf16 v16bf;
typedef __attribute__((ext_vector_type(8)))  __bf16 v8bf;
typedef __attribute__((ext_vector_type(8)))  float  v8f;

// ---------- helpers ----------
__device__ __forceinline__ __bf16 f2bf(float f) {
    union { float f; unsigned u; } v; v.f = f;
    unsigned r = v.u + 0x7FFFu + ((v.u >> 16) & 1u);   // round-to-nearest-even
    unsigned short h = (unsigned short)(r >> 16);
    return __builtin_bit_cast(__bf16, h);
}

__device__ __forceinline__ v8f vzero() {
    v8f z;
#pragma unroll
    for (int i = 0; i < 8; ++i) z[i] = 0.0f;
    return z;
}

__device__ __forceinline__ v8f wmma_bf16(v16bf a, v16bf b, v8f c) {
    // D = A(16x32) * B(32x16) + C(16x16), fp32 accumulate
    return __builtin_amdgcn_wmma_f32_16x16x32_bf16(false, a, false, b, (short)0, c,
                                                   false, false);
}

// A/B fragment loader from a bf16 row-major [rows, ld] matrix.
// Per ISA 16-bit A layout: lane holds K chunks {k0+(lane>=16?8:0)+[0..7]} and same +16.
__device__ __forceinline__ v16bf ldfrag_bf(const __bf16* __restrict__ base, int row,
                                           int ld, int k0, int lane) {
    const __bf16* p = base + (size_t)row * ld + k0 + ((lane & 16) >> 1);
    v8bf lo = *(const v8bf*)(p);
    v8bf hi = *(const v8bf*)(p + 16);
    return __builtin_shufflevector(lo, hi, 0, 1, 2, 3, 4, 5, 6, 7,
                                   8, 9, 10, 11, 12, 13, 14, 15);
}

// Same fragment, but from an fp32 source (convert in-register).
__device__ __forceinline__ v16bf ldfrag_f32(const float* __restrict__ base, int row,
                                            int ld, int k0, int lane) {
    const float* p = base + (size_t)row * ld + k0 + ((lane & 16) >> 1);
    v16bf r;
#pragma unroll
    for (int i = 0; i < 8; ++i) { r[i] = f2bf(p[i]); r[i + 8] = f2bf(p[i + 16]); }
    return r;
}

__device__ __forceinline__ v16bf ldfrag_bf_guard(const __bf16* __restrict__ base, int row,
                                                 int rowmax, int ld, int k0, int lane) {
    if (row < rowmax) return ldfrag_bf(base, row, ld, k0, lane);
    v16bf z;
#pragma unroll
    for (int i = 0; i < 16; ++i) z[i] = f2bf(0.0f);
    return z;
}

// ---------- one-time prep kernels ----------
__global__ void cvt_f32_bf16(const float* __restrict__ src, __bf16* __restrict__ dst, int n) {
    int i = blockIdx.x * blockDim.x + threadIdx.x;
    if (i < n) dst[i] = f2bf(src[i]);
}

// mode 0: encoder tokens seq[b,t]; mode 1: teacher-forced decoder tokens (PAD, tgt[:-1])
__global__ void gather_embed(const int* __restrict__ seq, const float* __restrict__ emb,
                             __bf16* __restrict__ out, int mode) {
    int idx = blockIdx.x * blockDim.x + threadIdx.x;     // over MR*Hv
    if (idx >= MR * Hv) return;
    int m = idx >> 9;              // t*B + b
    int h = idx & (Hv - 1);
    int t = m >> 6, b = m & (Bv - 1);
    int tok;
    if (mode == 0) tok = seq[b * Tv + t];
    else           tok = (t == 0) ? PADTOK : seq[b * Tv + (t - 1)];
    out[idx] = f2bf(emb[(size_t)tok * Hv + h]);
}

// ---------- generic bf16 GEMM: C[M,N] (f32) = A[M,K] * W[N,K]^T + bias[n] ----------
__global__ void gemm_bias(const __bf16* __restrict__ A, const __bf16* __restrict__ W,
                          const float* __restrict__ bias, float* __restrict__ C,
                          int M, int N, int K) {
    int lane = threadIdx.x & 31;
    int job  = blockIdx.x * (blockDim.x >> 5) + (threadIdx.x >> 5);
    int Nt = N >> 4;
    int njobs = (M >> 4) * Nt;
    if (job >= njobs) return;
    int tm = (job / Nt) << 4;
    int tn = (job % Nt) << 4;
    v8f acc = vzero();
    for (int k0 = 0; k0 < K; k0 += 32) {
        v16bf a = ldfrag_bf(A, tm + (lane & 15), K, k0, lane);
        v16bf b = ldfrag_bf(W, tn + (lane & 15), K, k0, lane);
        acc = wmma_bf16(a, b, acc);
    }
    int c  = tn + (lane & 15);
    int r0 = tm + ((lane & 16) >> 1);
    float bs = bias ? bias[c] : 0.0f;
#pragma unroll
    for (int r = 0; r < 8; ++r) C[(size_t)(r0 + r) * N + c] = acc[r] + bs;
}

// ---------- persistent single-WGP GRU recurrence ----------
struct RecSet {
    const float*  G;    // [T*B, 3H] precomputed x@Wi^T + bi (rows t*B+b)
    const __bf16* Wh;   // [3H, H] bf16
    const float*  bh;   // [3H]
    float*        ys;   // [T, B, H] hidden history output
    float*        h;    // [B, H] scratch (fp32 state)
    float*        gh;   // [B, 3H] scratch
    int           rev;  // 1 = backward time order
};

__global__ __launch_bounds__(1024) void gru_rec(RecSet s0, RecSet s1, const float* hinit) {
    RecSet S = (blockIdx.x == 0) ? s0 : s1;
    const float* hi = (blockIdx.x == 0) ? hinit : nullptr;
    const int tid = threadIdx.x, wave = tid >> 5, lane = tid & 31;

    for (int i = tid; i < Bv * Hv; i += 1024) S.h[i] = hi ? hi[i] : 0.0f;
    __threadfence(); __syncthreads();

    for (int s = 0; s < Tv; ++s) {
        int tau = S.rev ? (Tv - 1 - s) : s;
        const float* Gs = S.G + (size_t)tau * Bv * G3;

        // gh[64,1536] = bf16(h) @ Wh^T   (4 x 96 = 384 tiles over 32 waves)
        for (int tile = wave; tile < 384; tile += 32) {
            int tm = (tile & 3) << 4;
            int tn = (tile >> 2) << 4;
            v8f acc = vzero();
            for (int k0 = 0; k0 < Hv; k0 += 32) {
                v16bf a = ldfrag_f32(S.h, tm + (lane & 15), Hv, k0, lane);
                v16bf b = ldfrag_bf(S.Wh, tn + (lane & 15), Hv, k0, lane);
                acc = wmma_bf16(a, b, acc);
            }
            int c = tn + (lane & 15);
            int r0 = tm + ((lane & 16) >> 1);
#pragma unroll
            for (int r = 0; r < 8; ++r) S.gh[(r0 + r) * G3 + c] = acc[r];
        }
        __threadfence(); __syncthreads();

        // fused GRU gates + state update + history write
        for (int i = tid; i < Bv * Hv; i += 1024) {
            int b = i >> 9, j = i & (Hv - 1);
            float ir = Gs[b * G3 + j];
            float iz = Gs[b * G3 + Hv + j];
            float in = Gs[b * G3 + 2 * Hv + j];
            float hr = S.gh[b * G3 + j]          + S.bh[j];
            float hz = S.gh[b * G3 + Hv + j]     + S.bh[Hv + j];
            float hn = S.gh[b * G3 + 2 * Hv + j] + S.bh[2 * Hv + j];
            float r = 1.0f / (1.0f + __expf(-(ir + hr)));
            float z = 1.0f / (1.0f + __expf(-(iz + hz)));
            float n = tanhf(in + r * hn);
            float hnew = (1.0f - z) * n + z * S.h[i];
            S.h[i] = hnew;
            S.ys[(size_t)tau * Bv * Hv + i] = hnew;
        }
        __threadfence(); __syncthreads();
    }
}

__global__ void add2(const float* __restrict__ a, const float* __restrict__ b,
                     float* __restrict__ c, int n) {
    int i = blockIdx.x * blockDim.x + threadIdx.x;
    if (i < n) c[i] = a[i] + b[i];
}

// ---------- batched attention energies: E[t,b,s] = h2[t,b,:].enc_out[s,b,:] ----------
__global__ void energies(const float* __restrict__ h2, const float* __restrict__ eo,
                         float* __restrict__ E) {
    int lane = threadIdx.x & 31;
    int b = blockIdx.y;
    int tm = (blockIdx.x & 3) << 4;
    int tn = (blockIdx.x >> 2) << 4;
    const float* Ab = h2 + b * Hv;   // row t stride B*H
    const float* Bb = eo + b * Hv;   // row s stride B*H
    v8f acc = vzero();
    for (int k0 = 0; k0 < Hv; k0 += 32) {
        v16bf a = ldfrag_f32(Ab, tm + (lane & 15), Bv * Hv, k0, lane);
        v16bf w = ldfrag_f32(Bb, tn + (lane & 15), Bv * Hv, k0, lane);
        acc = wmma_bf16(a, w, acc);
    }
    int s  = tn + (lane & 15);
    int t0 = tm + ((lane & 16) >> 1);
#pragma unroll
    for (int r = 0; r < 8; ++r) E[((t0 + r) * Bv + b) * Tv + s] = acc[r];
}

__global__ void softmax64(const float* __restrict__ E, float* __restrict__ W) {
    __shared__ float sh[Tv];
    int row = blockIdx.x, s = threadIdx.x;
    float v = E[row * Tv + s];
    sh[s] = v; __syncthreads();
    for (int off = 32; off > 0; off >>= 1) {
        if (s < off) sh[s] = fmaxf(sh[s], sh[s + off]);
        __syncthreads();
    }
    float mx = sh[0]; __syncthreads();
    float e = __expf(v - mx);
    sh[s] = e; __syncthreads();
    for (int off = 32; off > 0; off >>= 1) {
        if (s < off) sh[s] += sh[s + off];
        __syncthreads();
    }
    W[row * Tv + s] = e / sh[0];
}

__global__ void ctx_kernel(const float* __restrict__ Wsm, const float* __restrict__ eo,
                           float* __restrict__ ctx) {
    __shared__ float w[Tv];
    int row = blockIdx.x;            // t*B + b
    int b = row & (Bv - 1);
    int h = threadIdx.x;             // 0..511
    if (h < Tv) w[h] = Wsm[row * Tv + h];
    __syncthreads();
    float acc = 0.0f;
    const float* eb = eo + b * Hv + h;
#pragma unroll 4
    for (int s = 0; s < Tv; ++s) acc += w[s] * eb[(size_t)s * Bv * Hv];
    ctx[(size_t)row * Hv + h] = acc;
}

// ---------- concat layer: co = tanh([h2, ctx] @ cW^T + cb), bf16 out ----------
__global__ void concat_gemm(const float* __restrict__ h2, const float* __restrict__ ctx,
                            const __bf16* __restrict__ cW, const float* __restrict__ cb,
                            __bf16* __restrict__ co) {
    int lane = threadIdx.x & 31;
    int job  = blockIdx.x * (blockDim.x >> 5) + (threadIdx.x >> 5);
    int tn = (job & 31) << 4;        // 512/16 = 32 N tiles
    int tm = (job >> 5) << 4;        // 256 M tiles
    v8f acc = vzero();
    for (int k0 = 0; k0 < 2 * Hv; k0 += 32) {
        const float* src = (k0 < Hv) ? h2 : ctx;
        int ke = (k0 < Hv) ? k0 : (k0 - Hv);
        v16bf a = ldfrag_f32(src, tm + (lane & 15), Hv, ke, lane);
        v16bf b = ldfrag_bf(cW, tn + (lane & 15), 2 * Hv, k0, lane);
        acc = wmma_bf16(a, b, acc);
    }
    int c  = tn + (lane & 15);
    int r0 = tm + ((lane & 16) >> 1);
    float bs = cb[c];
#pragma unroll
    for (int r = 0; r < 8; ++r)
        co[(size_t)(r0 + r) * Hv + c] = f2bf(tanhf(acc[r] + bs));
}

// ---------- output projection: 64x64 per wave, N-guarded (V = 20001) ----------
__global__ void logits_gemm(const __bf16* __restrict__ co, const __bf16* __restrict__ oW,
                            const float* __restrict__ ob, float* __restrict__ out) {
    int lane = threadIdx.x & 31;
    int job  = blockIdx.x * (blockDim.x >> 5) + (threadIdx.x >> 5);
    const int NB = (Vv + 63) / 64;   // 313
    if (job >= (MR / 64) * NB) return;
    int m0 = (job / NB) * 64;
    int n0 = (job % NB) * 64;
    v8f acc[4][4];
#pragma unroll
    for (int i = 0; i < 4; ++i)
#pragma unroll
        for (int j = 0; j < 4; ++j) acc[i][j] = vzero();

    for (int k0 = 0; k0 < Hv; k0 += 32) {
        v16bf a[4];
#pragma unroll
        for (int i = 0; i < 4; ++i)
            a[i] = ldfrag_bf(co, m0 + i * 16 + (lane & 15), Hv, k0, lane);
#pragma unroll
        for (int j = 0; j < 4; ++j) {
            v16bf b = ldfrag_bf_guard(oW, n0 + j * 16 + (lane & 15), Vv, Hv, k0, lane);
#pragma unroll
            for (int i = 0; i < 4; ++i) acc[i][j] = wmma_bf16(a[i], b, acc[i][j]);
        }
    }
#pragma unroll
    for (int j = 0; j < 4; ++j) {
        int c = n0 + j * 16 + (lane & 15);
        if (c < Vv) {
            float bs = ob[c];
#pragma unroll
            for (int i = 0; i < 4; ++i) {
                int r0 = m0 + i * 16 + ((lane & 16) >> 1);
#pragma unroll
                for (int r = 0; r < 8; ++r)
                    out[(size_t)(r0 + r) * Vv + c] = acc[i][j][r] + bs;
            }
        }
    }
}

// ---------- host ----------
extern "C" void kernel_launch(void* const* d_in, const int* in_sizes, int n_in,
                              void* d_out, int out_size, void* d_ws, size_t ws_size,
                              hipStream_t stream) {
    (void)in_sizes; (void)n_in; (void)out_size; (void)ws_size;
    const int*   input_seq  = (const int*)d_in[0];
    const int*   target_seq = (const int*)d_in[2];
    const float* enc_emb    = (const float*)d_in[3];
    const float* eWi_f = (const float*)d_in[4],  *eWh_f = (const float*)d_in[5];
    const float* ebi_f = (const float*)d_in[6],  *ebh_f = (const float*)d_in[7];
    const float* eWi_b = (const float*)d_in[8],  *eWh_b = (const float*)d_in[9];
    const float* ebi_b = (const float*)d_in[10], *ebh_b = (const float*)d_in[11];
    const float* dec_emb = (const float*)d_in[12];
    const float* dWi = (const float*)d_in[13], *dWh = (const float*)d_in[14];
    const float* dbi = (const float*)d_in[15], *dbh = (const float*)d_in[16];
    const float* cW = (const float*)d_in[17],  *cb = (const float*)d_in[18];
    const float* oW = (const float*)d_in[19],  *ob = (const float*)d_in[20];
    float* out = (float*)d_out;

    char* ws = (char*)d_ws;
    size_t off = 0;
    auto carve = [&](size_t bytes) -> void* {
        void* p = ws + off;
        off = (off + bytes + 255) & ~(size_t)255;
        return p;
    };
    __bf16* XS  = (__bf16*)carve((size_t)MR * Hv * 2);
    __bf16* XD  = (__bf16*)carve((size_t)MR * Hv * 2);
    __bf16* wIf = (__bf16*)carve((size_t)G3 * Hv * 2);
    __bf16* wHf = (__bf16*)carve((size_t)G3 * Hv * 2);
    __bf16* wIb = (__bf16*)carve((size_t)G3 * Hv * 2);
    __bf16* wHb = (__bf16*)carve((size_t)G3 * Hv * 2);
    __bf16* wDi = (__bf16*)carve((size_t)G3 * Hv * 2);
    __bf16* wDh = (__bf16*)carve((size_t)G3 * Hv * 2);
    __bf16* wC  = (__bf16*)carve((size_t)Hv * 2 * Hv * 2);
    __bf16* wO  = (__bf16*)carve((size_t)Vv * Hv * 2);
    float* Gf  = (float*)carve((size_t)MR * G3 * 4);
    float* Gb  = (float*)carve((size_t)MR * G3 * 4);
    float* Gd  = (float*)carve((size_t)MR * G3 * 4);
    float* ysF = (float*)carve((size_t)MR * Hv * 4);
    float* ysB = (float*)carve((size_t)MR * Hv * 4);
    float* eo  = (float*)carve((size_t)MR * Hv * 4);
    float* h2  = (float*)carve((size_t)MR * Hv * 4);
    float* ctx = (float*)carve((size_t)MR * Hv * 4);
    float* hb0 = (float*)carve((size_t)Bv * Hv * 4);
    float* hb1 = (float*)carve((size_t)Bv * Hv * 4);
    float* hb2 = (float*)carve((size_t)Bv * Hv * 4);
    float* gh0 = (float*)carve((size_t)Bv * G3 * 4);
    float* gh1 = (float*)carve((size_t)Bv * G3 * 4);
    float* gh2 = (float*)carve((size_t)Bv * G3 * 4);
    float* E   = (float*)carve((size_t)Tv * Bv * Tv * 4);
    float* Wsm = (float*)carve((size_t)Tv * Bv * Tv * 4);
    __bf16* co = (__bf16*)carve((size_t)MR * Hv * 2);

    // 1) convert weights to bf16 (out_W / Wi / Wh / concat_W)
    auto cvt = [&](const float* s, __bf16* d, int n) {
        cvt_f32_bf16<<<(n + 255) / 256, 256, 0, stream>>>(s, d, n);
    };
    cvt(eWi_f, wIf, G3 * Hv); cvt(eWh_f, wHf, G3 * Hv);
    cvt(eWi_b, wIb, G3 * Hv); cvt(eWh_b, wHb, G3 * Hv);
    cvt(dWi,   wDi, G3 * Hv); cvt(dWh,   wDh, G3 * Hv);
    cvt(cW,    wC,  Hv * 2 * Hv); cvt(oW, wO, Vv * Hv);

    // 2) embedding gathers
    gather_embed<<<(MR * Hv + 255) / 256, 256, 0, stream>>>(input_seq,  enc_emb, XS, 0);
    gather_embed<<<(MR * Hv + 255) / 256, 256, 0, stream>>>(target_seq, dec_emb, XD, 1);

    // 3) batched input-gate GEMMs  G = X @ Wi^T + bi  (M=4096, N=1536, K=512)
    {
        int jobs = (MR / 16) * (G3 / 16);
        int blocks = (jobs + 7) / 8;
        gemm_bias<<<blocks, 256, 0, stream>>>(XS, wIf, ebi_f, Gf, MR, G3, Hv);
        gemm_bias<<<blocks, 256, 0, stream>>>(XS, wIb, ebi_b, Gb, MR, G3, Hv);
        gemm_bias<<<blocks, 256, 0, stream>>>(XD, wDi, dbi,  Gd, MR, G3, Hv);
    }

    // 4) encoder: fwd + bwd recurrences concurrently (one WGP each)
    RecSet sf { Gf, wHf, ebh_f, ysF, hb0, gh0, 0 };
    RecSet sb { Gb, wHb, ebh_b, ysB, hb1, gh1, 1 };
    gru_rec<<<2, 1024, 0, stream>>>(sf, sb, nullptr);

    // 5) decoder recurrence, h0 = final forward hidden = ysF[T-1]
    RecSet sd { Gd, wDh, dbh, h2, hb2, gh2, 0 };
    gru_rec<<<1, 1024, 0, stream>>>(sd, sd, ysF + (size_t)(Tv - 1) * Bv * Hv);

    // 6) enc_out = ysF + ysB (reversal already folded into store indices)
    add2<<<(MR * Hv + 255) / 256, 256, 0, stream>>>(ysF, ysB, eo, MR * Hv);

    // 7) attention energies (batched over b), softmax, context
    energies<<<dim3(16, Bv), 32, 0, stream>>>(h2, eo, E);
    softmax64<<<MR, Tv, 0, stream>>>(E, Wsm);
    ctx_kernel<<<MR, Hv, 0, stream>>>(Wsm, eo, ctx);

    // 8) concat layer (K=1024 two-source A), then output projection
    concat_gemm<<<(MR / 16) * (Hv / 16) / 8, 256, 0, stream>>>(h2, ctx, wC, cb, co);
    {
        int jobs = (MR / 64) * ((Vv + 63) / 64);   // 64 * 313
        logits_gemm<<<(jobs + 7) / 8, 256, 0, stream>>>(co, wO, ob, out);
    }
}
